// Minerva2_73778948211272
// MI455X (gfx1250) — compile-verified, matrix operand order checked by reading, not run on hardware
//
#include <hip/hip_runtime.h>

typedef __attribute__((ext_vector_type(16))) __bf16 bf16x16;
typedef __attribute__((ext_vector_type(8)))  __bf16 bf16x8;
typedef __attribute__((ext_vector_type(2)))  __bf16 bf16x2;
typedef __attribute__((ext_vector_type(8)))  float  f32x8;
typedef unsigned int  v4u __attribute__((ext_vector_type(4)));
typedef int           v8i __attribute__((ext_vector_type(8)));
typedef int           v4i __attribute__((ext_vector_type(4)));
typedef unsigned short u16;
typedef __attribute__((address_space(3))) u16 lds_u16;

#define ATHENA_USE_TDM 1

#define DIN  1024
#define DREP 1024
#define NX   4096
#define ND   4096
#define NB   4
#define MTOT (NB * NX)   // 16384 rows for both X and D views

// round-to-nearest-even f32 -> bf16 bit pattern
__device__ __forceinline__ u16 f32_bf16(float f) {
    union { float f; unsigned u; } v; v.f = f;
    unsigned r = v.u + 0x7FFFu + ((v.u >> 16) & 1u);
    return (u16)(r >> 16);
}

// pack two f32 -> packed bf16 pair (v_cvt_pk_bf16_f32 when available)
__device__ __forceinline__ unsigned pack_bf16(float x, float y) {
#if __has_builtin(__builtin_amdgcn_cvt_pk_bf16_f32)
    union { bf16x2 v; unsigned u; } c;
    c.v = __builtin_amdgcn_cvt_pk_bf16_f32(x, y);
    return c.u;
#else
    return (unsigned)f32_bf16(x) | ((unsigned)f32_bf16(y) << 16);
#endif
}

// Load a 16x32 bf16 WMMA operand fragment from LDS.
// Layout (ISA 7.12.2, 16-bit A 16x32): lane<16 holds K {0..7,16..23},
// lane>=16 holds K {8..15,24..31}; row = lane&15.
__device__ __forceinline__ bf16x16 ldsFrag(const u16* lds, int row0, int ld,
                                           int k0, int lane) {
    int r  = row0 + (lane & 15);
    int kb = k0 + ((lane & 16) ? 8 : 0);
    const bf16x8* p = (const bf16x8*)(lds + (size_t)r * ld + kb);
    bf16x8 lo = p[0];      // K = kb .. kb+7
    bf16x8 hi = p[2];      // K = kb+16 .. kb+23
    bf16x16 o;
#pragma unroll
    for (int i = 0; i < 8; ++i) { o[i] = lo[i]; o[i + 8] = hi[i]; }
    return o;
}

#if ATHENA_USE_TDM
// One Tensor-Data-Mover 2D tile load: tile_h rows x tile_w elements (2 bytes
// each), global row stride `stride` elements, into LDS at lds_addr.
// D# layout per CDNA5 ISA ch.8: group0 = {count, lds_addr, global_addr, type=2},
// group1 = {data_size, tensor dims, tile dims, 48-bit dim0 stride}.
// This toolchain's builtin is the 6-arg form:
//   (uint32x4 g0, int32x8 g1, int32x4 g2, int32x4 g3, int32x8, int cpol)
__device__ __forceinline__ void tdm_load_2d(unsigned lds_addr, const void* gaddr,
                                            unsigned tile_w, unsigned tile_h,
                                            unsigned tensor_w, unsigned tensor_h,
                                            unsigned stride) {
    unsigned long long ga = (unsigned long long)gaddr;
    v4u g0;
    g0[0] = 1u;                                              // count=1 (valid), user mode
    g0[1] = lds_addr;                                        // LDS byte address
    g0[2] = (unsigned)(ga & 0xFFFFFFFFu);                    // global_addr[31:0]
    g0[3] = (unsigned)((ga >> 32) & 0x1FFFFFFu) | (2u << 30); // addr[56:32] | type=2
    v8i g1;
    g1[0] = (int)(1u << 16);                                 // data_size=1 (2B), mask=0
    g1[1] = (int)((tensor_w & 0xFFFFu) << 16);               // tensor_dim0[15:0]
    g1[2] = (int)(((tensor_w >> 16) & 0xFFFFu) | ((tensor_h & 0xFFFFu) << 16));
    g1[3] = (int)(((tensor_h >> 16) & 0xFFFFu) | (tile_w << 16)); // tile_dim0
    g1[4] = (int)(tile_h & 0xFFFFu);                         // tile_dim1, tile_dim2=0
    g1[5] = (int)stride;                                     // dim0_stride[31:0]
    g1[6] = 0;                                               // dim0_stride[47:32], dim1_stride lo
    g1[7] = 0;
    v4i z4 = {0, 0, 0, 0};
    v8i z8 = {0, 0, 0, 0, 0, 0, 0, 0};
    __builtin_amdgcn_tensor_load_to_lds(g0, g1, z4, z4, z8, 0);
}
#endif

// ---------------------------------------------------------------------------
// Projection GEMM: Y[m,r] = sum_i In[m,i]*W[r,i] + bias[r], Y stored bf16.
// Block tile 128(M) x 64(N), 8 waves in 2(M) x 4(N); each wave: 64x16 -> 4 WMMA.
// Global f32 loads for step k+1 are software-pipelined above the WMMAs of step k.
// ---------------------------------------------------------------------------
__global__ __launch_bounds__(256)
void proj_kernel(const float* __restrict__ In, const float* __restrict__ W,
                 const float* __restrict__ bias, u16* __restrict__ Yw) {
    __shared__ __align__(16) u16 ldsA[128 * 32];
    __shared__ __align__(16) u16 ldsB[64 * 32];

    const int tid  = threadIdx.x;
    const int lane = tid & 31;
    const int wave = tid >> 5;
    const int wm   = wave >> 2;            // 0..1
    const int wn   = wave & 3;             // 0..3
    const int m0   = blockIdx.y * 128;
    const int n0   = blockIdx.x * 64;

    f32x8 acc[4];
#pragma unroll
    for (int s = 0; s < 4; ++s)
#pragma unroll
        for (int i = 0; i < 8; ++i) acc[s][i] = 0.0f;

    const int c4    = (tid & 7) * 4;       // 0,4,..,28
    const int rbase = tid >> 3;            // 0..31

    float4 ra[4], rw[2];
    auto loadRegs = [&](int k0) {
#pragma unroll
        for (int it = 0; it < 4; ++it)
            ra[it] = *(const float4*)(In + (size_t)(m0 + rbase + it * 32) * DIN + k0 + c4);
#pragma unroll
        for (int it = 0; it < 2; ++it)
            rw[it] = *(const float4*)(W + (size_t)(n0 + rbase + it * 32) * DIN + k0 + c4);
    };

    loadRegs(0);
    for (int ks = 0; ks < DIN / 32; ++ks) {
        __syncthreads();
        // convert + store staged registers into LDS
#pragma unroll
        for (int it = 0; it < 4; ++it)
            *(uint2*)(ldsA + (rbase + it * 32) * 32 + c4) =
                make_uint2(pack_bf16(ra[it].x, ra[it].y), pack_bf16(ra[it].z, ra[it].w));
#pragma unroll
        for (int it = 0; it < 2; ++it)
            *(uint2*)(ldsB + (rbase + it * 32) * 32 + c4) =
                make_uint2(pack_bf16(rw[it].x, rw[it].y), pack_bf16(rw[it].z, rw[it].w));
        __syncthreads();
        if (ks + 1 < DIN / 32) loadRegs((ks + 1) * 32);   // pipeline next step's globals

        bf16x16 bfr = ldsFrag(ldsB, wn * 16, 32, 0, lane);
#pragma unroll
        for (int s = 0; s < 4; ++s) {
            bf16x16 afr = ldsFrag(ldsA, wm * 64 + s * 16, 32, 0, lane);
            acc[s] = __builtin_amdgcn_wmma_f32_16x16x32_bf16(
                false, afr, false, bfr, (short)0, acc[s], false, false);
        }
    }

    // epilogue: +bias, f32 -> bf16 (C layout: N=lane&15, M=v+8*(lane>=16))
    const int   ncol = n0 + wn * 16 + (lane & 15);
    const float bv   = bias[ncol];
    const int   moff = (lane & 16) ? 8 : 0;
#pragma unroll
    for (int s = 0; s < 4; ++s)
#pragma unroll
        for (int v = 0; v < 8; ++v) {
            int m = m0 + wm * 64 + s * 16 + v + moff;
            Yw[(size_t)m * DREP + ncol] = f32_bf16(acc[s][v] + bv);
        }
}

// ---------------------------------------------------------------------------
// Fused: echo[b,x] += sum_d (Xw[b,x,:] . Dw[b,d,:])^3 * R[b,d]
// Block: 64 x-rows, 512-wide Nd slice as 4 d-tiles of 128.
// Xw 64x1024 bf16 tile (128 KB) pinned in LDS; Dw tiles streamed by the
// Tensor Data Mover (one TENSOR_LOAD_TO_LDS per 8 KB tile, TENSORcnt-tracked,
// issued by wave 0 only) into a double buffer. Waves spend issue slots on WMMA.
// Wave tile 32x32 -> 4 WMMA per k-step from 2 A + 2 B fragments.
// ---------------------------------------------------------------------------
__global__ __launch_bounds__(256)
void fused_kernel(const u16* __restrict__ Xw, const u16* __restrict__ Dw,
                  const float* __restrict__ R, float* __restrict__ echo) {
    __shared__ __align__(16) u16 ldsX[64 * 1024];       // 128 KB persistent A tile
    __shared__ __align__(16) u16 ldsD[2][128 * 32];     // 2 x 8 KB streamed B tiles
    __shared__ float eacc[64];

    const int tid  = threadIdx.x;
    const int lane = tid & 31;
    const int wave = tid >> 5;
    const int wm   = wave >> 2;            // 0..1 -> 32-row half (M)
    const int wn   = wave & 3;             // 0..3 -> 32-col strip (N=d)
    const int b    = blockIdx.z;
    const int x0   = blockIdx.y * 64;
    const int d0   = blockIdx.x * 512;

    if (tid < 64) eacc[tid] = 0.0f;

#if ATHENA_USE_TDM
    auto ldsByteAddr = [](const u16* p) {
        return (unsigned)(unsigned long long)(lds_u16*)p;
    };
    auto stageD = [&](int dbase, int k0, int buf) {   // uniform; call from wave 0
        tdm_load_2d(ldsByteAddr(ldsD[buf]),
                    Dw + ((size_t)b * ND + dbase) * DREP + k0,
                    /*tile_w=*/32, /*tile_h=*/128,
                    /*tensor_w=*/32, /*tensor_h=*/128, /*stride=*/DREP);
    };
    if (wave == 0) {
        // pin Xw tile: one TDM op for the whole 64x1024 bf16 tile
        tdm_load_2d(ldsByteAddr(ldsX), Xw + ((size_t)b * NX + x0) * DREP,
                    /*tile_w=*/1024, /*tile_h=*/64,
                    /*tensor_w=*/1024, /*tensor_h=*/64, /*stride=*/DREP);
    }
#else
    {   // pin Xw tile: flat b128 copy
        const uint4* src = (const uint4*)(Xw + ((size_t)b * NX + x0) * DREP);
        uint4* dst = (uint4*)ldsX;
        for (int i = tid; i < (64 * 1024) / 8; i += 256) dst[i] = src[i];
    }
    __syncthreads();
    const int drow = tid >> 1;             // 0..127
    const int dch  = (tid & 1) * 16;       // 0 or 16 halves
    auto stageD = [&](int dbase, int k0, int buf) {
        const u16* g = Dw + ((size_t)b * ND + dbase + drow) * DREP + k0 + dch;
        lds_u16*   l = (lds_u16*)(ldsD[buf] + drow * 32 + dch);
        asm volatile("global_load_async_to_lds_b128 %0, %1, off"
                     :: "v"(l), "v"(g) : "memory");
        asm volatile("global_load_async_to_lds_b128 %0, %1, off offset:16"
                     :: "v"(l), "v"(g) : "memory");
    };
#endif

    for (int dt = 0; dt < 4; ++dt) {
        const int dbase = d0 + dt * 128;

        f32x8 acc[2][2];
#pragma unroll
        for (int s = 0; s < 2; ++s)
#pragma unroll
            for (int t = 0; t < 2; ++t)
#pragma unroll
                for (int i = 0; i < 8; ++i) acc[s][t][i] = 0.0f;

#if ATHENA_USE_TDM
        if (wave == 0) stageD(dbase, 0, 0);
#else
        stageD(dbase, 0, 0);
#endif
        for (int ks = 0; ks < DREP / 32; ++ks) {
            const int k0  = ks * 32;
            const int buf = ks & 1;
#if ATHENA_USE_TDM
            if (wave == 0) __builtin_amdgcn_s_wait_tensorcnt(0); // tiles landed
            __syncthreads();
            if (wave == 0 && ks + 1 < DREP / 32)
                stageD(dbase, k0 + 32, buf ^ 1);                 // overlap next tile
#else
            asm volatile("s_wait_asynccnt 0x0" ::: "memory");
            __syncthreads();
            if (ks + 1 < DREP / 32) stageD(dbase, k0 + 32, buf ^ 1);
#endif

            bf16x16 bfr0 = ldsFrag(ldsD[buf], wn * 32, 32, 0, lane);
            bf16x16 bfr1 = ldsFrag(ldsD[buf], wn * 32 + 16, 32, 0, lane);
#pragma unroll
            for (int s = 0; s < 2; ++s) {
                bf16x16 afr = ldsFrag(ldsX, wm * 32 + s * 16, 1024, k0, lane);
                acc[s][0] = __builtin_amdgcn_wmma_f32_16x16x32_bf16(
                    false, afr, false, bfr0, (short)0, acc[s][0], false, false);
                acc[s][1] = __builtin_amdgcn_wmma_f32_16x16x32_bf16(
                    false, afr, false, bfr1, (short)0, acc[s][1], false, false);
            }
        }

        // epilogue: cube, scale by R[d], reduce rows via LDS float atomics
        const int moff = (lane & 16) ? 8 : 0;
#pragma unroll
        for (int t = 0; t < 2; ++t) {
            const int   nloc = wn * 32 + t * 16 + (lane & 15);
            const float rv   = R[(size_t)b * ND + dbase + nloc];
#pragma unroll
            for (int s = 0; s < 2; ++s)
#pragma unroll
                for (int v = 0; v < 8; ++v) {
                    float a = acc[s][t][v];
                    atomicAdd(&eacc[wm * 32 + s * 16 + v + moff], a * a * a * rv);
                }
        }
    }
    __syncthreads();
    if (tid < 64) atomicAdd(&echo[(size_t)b * NX + x0 + tid], eacc[tid]);
}

__global__ __launch_bounds__(256)
void zero_kernel(float* __restrict__ p, int n) {
    int i = blockIdx.x * 256 + threadIdx.x;
    if (i < n) p[i] = 0.0f;
}

__global__ __launch_bounds__(256)
void finalize_kernel(const float* __restrict__ echo, const float* __restrict__ wr_w,
                     const float* __restrict__ wr_b, float* __restrict__ out, int n) {
    int i = blockIdx.x * 256 + threadIdx.x;
    if (i < n) out[i] = echo[i] * wr_w[0] + wr_b[0];
}

extern "C" void kernel_launch(void* const* d_in, const int* in_sizes, int n_in,
                              void* d_out, int out_size, void* d_ws, size_t ws_size,
                              hipStream_t stream) {
    const float* X    = (const float*)d_in[0];
    const float* D    = (const float*)d_in[1];
    const float* R    = (const float*)d_in[2];
    const float* Wx_w = (const float*)d_in[3];
    const float* Wx_b = (const float*)d_in[4];
    const float* Wd_w = (const float*)d_in[5];
    const float* Wd_b = (const float*)d_in[6];
    const float* Wr_w = (const float*)d_in[7];
    const float* Wr_b = (const float*)d_in[8];
    float* out = (float*)d_out;

    // workspace: Xw bf16 (32MB) | Dw bf16 (32MB) | echo f32 (64KB)
    u16*   Xw   = (u16*)d_ws;
    u16*   Dw   = Xw + (size_t)MTOT * DREP;
    float* echo = (float*)(Dw + (size_t)MTOT * DREP);

    zero_kernel<<<dim3((MTOT + 255) / 256), 256, 0, stream>>>(echo, MTOT);

    dim3 pgrid(DREP / 64, MTOT / 128);     // (16, 128)
    proj_kernel<<<pgrid, 256, 0, stream>>>(X, Wx_w, Wx_b, Xw);
    proj_kernel<<<pgrid, 256, 0, stream>>>(D, Wd_w, Wd_b, Dw);

    dim3 fgrid(ND / 512, NX / 64, NB);     // (8, 64, 4) = 2048 blocks
    fused_kernel<<<fgrid, 256, 0, stream>>>(Xw, Dw, R, echo);

    finalize_kernel<<<dim3((MTOT + 255) / 256), 256, 0, stream>>>(echo, Wr_w, Wr_b, out, MTOT);
}
